// FrameTransformerEncoder_6811818131913
// MI455X (gfx1250) — matrix operand order repaired and checked by reading, not compiled, not run on hardware
//
#include <hip/hip_runtime.h>
#include <hip/hip_bf16.h>
#include <math.h>

// Problem constants (from reference)
#define Bc   2
#define CINc 2
#define Cc   8
#define Fc   1024
#define Wd   512
#define Hc   8
#define HD   128
#define CEXP 32   // COUT * EXP

typedef __attribute__((ext_vector_type(16))) _Float16 v16h;
typedef __attribute__((ext_vector_type(8)))  float    v8f;
typedef __attribute__((ext_vector_type(2)))  _Float16 h2;

union CFrag { v8f v; float e[8]; };

__device__ __forceinline__ v8f wmma16x16x32(v16h a, v16h b, v8f c) {
    return __builtin_amdgcn_wmma_f32_16x16x32_f16(
        false, a, false, b, (short)0, c, false, false);
}

// ---------------------------------------------------------------------------
// 1) embed: 1x1 conv CIN->COUT, plus copy x into concat output channels 0..1
// ---------------------------------------------------------------------------
__global__ void embed_kernel(const float* __restrict__ x,
                             const float* __restrict__ ew,
                             const float* __restrict__ eb,
                             float* __restrict__ h,
                             float* __restrict__ out_concat) {
    int idx = blockIdx.x * blockDim.x + threadIdx.x;       // over B*F*W
    if (idx >= Bc * Fc * Wd) return;
    int w = idx % Wd;
    int f = (idx / Wd) % Fc;
    int b = idx / (Wd * Fc);
    float x0 = x[((size_t)(b * CINc + 0) * Fc + f) * Wd + w];
    float x1 = x[((size_t)(b * CINc + 1) * Fc + f) * Wd + w];
    out_concat[((size_t)(b * (CINc + Cc) + 0) * Fc + f) * Wd + w] = x0;
    out_concat[((size_t)(b * (CINc + Cc) + 1) * Fc + f) * Wd + w] = x1;
#pragma unroll
    for (int co = 0; co < Cc; ++co) {
        float v = x0 * ew[co * CINc + 0] + x1 * ew[co * CINc + 1] + eb[co];
        h[((size_t)(b * Cc + co) * Fc + f) * Wd + w] = v;
    }
}

// ---------------------------------------------------------------------------
// 2) layernorm over F axis. One block (256 thr) per (b,c,w) column.
// ---------------------------------------------------------------------------
__global__ void layernorm_kernel(const float* __restrict__ in,
                                 const float* __restrict__ g,
                                 const float* __restrict__ be,
                                 float* __restrict__ out) {
    int col = blockIdx.x;                    // b*C*W columns
    int w = col % Wd;
    int c = (col / Wd) % Cc;
    int b = col / (Wd * Cc);
    const float* base = in + ((size_t)(b * Cc + c)) * Fc * Wd + w;
    float* ob = out + ((size_t)(b * Cc + c)) * Fc * Wd + w;
    __shared__ float red[256];

    float s = 0.f;
    for (int f = threadIdx.x; f < Fc; f += 256) s += base[(size_t)f * Wd];
    red[threadIdx.x] = s; __syncthreads();
    for (int o = 128; o > 0; o >>= 1) {
        if (threadIdx.x < o) red[threadIdx.x] += red[threadIdx.x + o];
        __syncthreads();
    }
    float mu = red[0] * (1.f / Fc);
    __syncthreads();

    float s2 = 0.f;
    for (int f = threadIdx.x; f < Fc; f += 256) {
        float d = base[(size_t)f * Wd] - mu; s2 += d * d;
    }
    red[threadIdx.x] = s2; __syncthreads();
    for (int o = 128; o > 0; o >>= 1) {
        if (threadIdx.x < o) red[threadIdx.x] += red[threadIdx.x + o];
        __syncthreads();
    }
    float rstd = rsqrtf(red[0] * (1.f / Fc) + 1e-8f);

    for (int f = threadIdx.x; f < Fc; f += 256) {
        float xn = (base[(size_t)f * Wd] - mu) * rstd;
        ob[(size_t)f * Wd] = xn * g[c * Fc + f] + be[c * Fc + f];
    }
}

// ---------------------------------------------------------------------------
// 3) depthwise 9x9 conv, pad 4
// ---------------------------------------------------------------------------
__global__ void dwconv9_kernel(const float* __restrict__ in,
                               const float* __restrict__ wt,
                               float* __restrict__ out) {
    size_t idx = (size_t)blockIdx.x * blockDim.x + threadIdx.x;
    if (idx >= (size_t)Bc * Cc * Fc * Wd) return;
    int w = idx % Wd;
    int f = (idx / Wd) % Fc;
    int c = (idx / ((size_t)Wd * Fc)) % Cc;
    int b = idx / ((size_t)Wd * Fc * Cc);
    const float* ib = in + ((size_t)(b * Cc + c)) * Fc * Wd;
    const float* wc = wt + c * 81;
    float acc = 0.f;
    for (int i = 0; i < 9; ++i) {
        int ff = f - 4 + i;
        if (ff < 0 || ff >= Fc) continue;
#pragma unroll
        for (int j = 0; j < 9; ++j) {
            int ww = w - 4 + j;
            if (ww < 0 || ww >= Wd) continue;
            acc += ib[(size_t)ff * Wd + ww] * wc[i * 9 + j];
        }
    }
    out[idx] = acc;
}

// ===========================================================================
// WMMA GEMM kernels. Block = 256 thr (8 waves, 4(m) x 2(n) wave grid).
// Block tile 64(M) x 64(N), staged K-depth 64 (2 WMMA K-chunks).
// LDS is kept in fragment-resident (swizzled) layout:
//   A: As2[mt(4)][lane(32)][e(32)]  -- lane L holds row (L&15) of subtile mt,
//      K split per CDNA5 A-layout: lanes 0-15 K%16<8-group, lanes 16-31 other.
//   B: Bs2[nt(4)][lane(32)][e(32)]  -- lane L holds col (L&15), K split same as A.
// Fragment load = one 32B vector load (2x ds_load_b128).
// Inverse maps used at stage-in time (f32 pair -> packed 2xf16 ds_store_b32):
//   A elem (i,k): lane=(i&15)+16*((k>>3)&1), e=(k>>5)*16+(k&7)+8*((k>>4)&1)
//   B elem (k,n): lane=(n&15)+16*((k>>3)&1), e=(k>>5)*16+(k&7)+8*((k>>4)&1)
// ===========================================================================

// ---------------------------------------------------------------------------
// 4) mc_linear: Y[b,c,fo,w] = sum_fi Wlin[c,fo,fi] * X[b,c,fi,w]
//    grid = (F/64, W/64, B*C)
// ---------------------------------------------------------------------------
__global__ __launch_bounds__(256) void mclinear_gemm_kernel(
        const float* __restrict__ Wlin, const float* __restrict__ X,
        float* __restrict__ Y) {
    int bc = blockIdx.z;
    int c  = bc % Cc;
    int m0 = blockIdx.x * 64;
    int n0 = blockIdx.y * 64;
    const float* A  = Wlin + (size_t)c * Fc * Fc;
    const float* Xb = X + (size_t)bc * Fc * Wd;
    float*       Yb = Y + (size_t)bc * Fc * Wd;

    __shared__ __align__(32) _Float16 As2[4096];
    __shared__ __align__(32) _Float16 Bs2[4096];

    int tid = threadIdx.x, lane = tid & 31, wid = tid >> 5;
    int mt = wid & 3, ntB = (wid >> 2) * 2;

    CFrag acc0, acc1;
#pragma unroll
    for (int r = 0; r < 8; ++r) { acc0.e[r] = 0.f; acc1.e[r] = 0.f; }

    for (int k0 = 0; k0 < Fc; k0 += 64) {
        if (k0 + 64 < Fc)
            __builtin_prefetch(&A[(size_t)(m0 + wid) * Fc + k0 + 64], 0, 1);
        // stage A: rows m, contiguous k pairs
        for (int t = tid; t < 2048; t += 256) {
            int i = t >> 5, jp = (t & 31) * 2;
            float a0 = A[(size_t)(m0 + i) * Fc + (k0 + jp)];
            float a1 = A[(size_t)(m0 + i) * Fc + (k0 + jp + 1)];
            int lne = (i & 15) + 16 * ((jp >> 3) & 1);
            int e   = (jp >> 5) * 16 + (jp & 7) + 8 * ((jp >> 4) & 1);
            h2 p; p.x = (_Float16)a0; p.y = (_Float16)a1;
            *(h2*)&As2[(((i >> 4) * 32 + lne) * 32) + e] = p;
        }
        // stage B: k-strided pairs, n contiguous across threads
        for (int t = tid; t < 2048; t += 256) {
            int kp = (t >> 6) * 2, n = t & 63;
            float b0 = Xb[(size_t)(k0 + kp) * Wd + (n0 + n)];
            float b1 = Xb[(size_t)(k0 + kp + 1) * Wd + (n0 + n)];
            int lne = (n & 15) + 16 * ((kp >> 3) & 1);
            int e   = (kp >> 5) * 16 + (kp & 7) + 8 * ((kp >> 4) & 1);
            h2 p; p.x = (_Float16)b0; p.y = (_Float16)b1;
            *(h2*)&Bs2[(((n >> 4) * 32 + lne) * 32) + e] = p;
        }
        __syncthreads();
#pragma unroll
        for (int ck = 0; ck < 2; ++ck) {
            v16h af  = *(const v16h*)&As2[((mt * 32 + lane) * 32) + ck * 16];
            v16h bf0 = *(const v16h*)&Bs2[((ntB * 32 + lane) * 32) + ck * 16];
            v16h bf1 = *(const v16h*)&Bs2[(((ntB + 1) * 32 + lane) * 32) + ck * 16];
            acc0.v = wmma16x16x32(af, bf0, acc0.v);
            acc1.v = wmma16x16x32(af, bf1, acc1.v);
        }
        __syncthreads();
    }

    int ncol  = lane & 15;
    int mBase = m0 + mt * 16 + ((lane >> 4) ? 8 : 0);
    int nOut0 = n0 + ntB * 16 + ncol;
    int nOut1 = n0 + (ntB + 1) * 16 + ncol;
#pragma unroll
    for (int r = 0; r < 8; ++r) {
        Yb[(size_t)(mBase + r) * Wd + nOut0] = acc0.e[r];
        Yb[(size_t)(mBase + r) * Wd + nOut1] = acc1.e[r];
    }
}

// ---------------------------------------------------------------------------
// 5) rotary embedding in-place (pairs; pos = w, dim pair within head)
// ---------------------------------------------------------------------------
__global__ void rotary_kernel(float* __restrict__ q) {
    size_t idx = (size_t)blockIdx.x * blockDim.x + threadIdx.x;
    if (idx >= (size_t)Bc * Cc * (Fc / 2) * Wd) return;
    int w  = idx % Wd;
    int fp = (idx / Wd) % (Fc / 2);
    int bc = idx / ((size_t)Wd * (Fc / 2));
    int f    = fp * 2;
    int half = (f % HD) >> 1;
    float inv = __expf(-(float)half * 0.1439156602f);   // 10000^(-half/64)
    float ang = (float)w * inv;
    float cs = __cosf(ang), sn = __sinf(ang);
    float* base = q + ((size_t)bc * Fc + f) * Wd + w;
    float x0 = base[0];
    float x1 = base[Wd];
    base[0]  = x0 * cs - x1 * sn;
    base[Wd] = x1 * cs + x0 * sn;
}

// ---------------------------------------------------------------------------
// 6) qk GEMM: s[wq,wk] = (1/32)*sum_d Q[d,wq]*K[d,wk] + prev_qk
//    Both operands K-major in memory (row = d, stride Wd). grid=(8,8,b*c*h)
// ---------------------------------------------------------------------------
__global__ __launch_bounds__(256) void qk_gemm_kernel(
        const float* __restrict__ Q, const float* __restrict__ K,
        const float* __restrict__ prevqk, float* __restrict__ QKout) {
    int z  = blockIdx.z;
    int h  = z % Hc;
    int bc = z / Hc;
    int m0 = blockIdx.x * 64;
    int n0 = blockIdx.y * 64;
    const float* Qb = Q + (size_t)bc * Fc * Wd + (size_t)h * HD * Wd;
    const float* Kb = K + (size_t)bc * Fc * Wd + (size_t)h * HD * Wd;
    const float* Pb = prevqk + (size_t)z * Wd * Wd;
    float*       Ob = QKout + (size_t)z * Wd * Wd;

    __shared__ __align__(32) _Float16 As2[4096];
    __shared__ __align__(32) _Float16 Bs2[4096];

    int tid = threadIdx.x, lane = tid & 31, wid = tid >> 5;
    int mt = wid & 3, ntB = (wid >> 2) * 2;

    CFrag acc0, acc1;
#pragma unroll
    for (int r = 0; r < 8; ++r) { acc0.e[r] = 0.f; acc1.e[r] = 0.f; }

    for (int k0 = 0; k0 < HD; k0 += 64) {
        // stage A: elem (m=i, k=jp) = Qb[k*Wd + m]; i contiguous across threads
        for (int t = tid; t < 2048; t += 256) {
            int jp = (t >> 6) * 2, i = t & 63;
            float a0 = Qb[(size_t)(k0 + jp) * Wd + (m0 + i)];
            float a1 = Qb[(size_t)(k0 + jp + 1) * Wd + (m0 + i)];
            int lne = (i & 15) + 16 * ((jp >> 3) & 1);
            int e   = (jp >> 5) * 16 + (jp & 7) + 8 * ((jp >> 4) & 1);
            h2 p; p.x = (_Float16)a0; p.y = (_Float16)a1;
            *(h2*)&As2[(((i >> 4) * 32 + lne) * 32) + e] = p;
        }
        // stage B: elem (k=jp, n) = Kb[k*Wd + n]
        for (int t = tid; t < 2048; t += 256) {
            int jp = (t >> 6) * 2, n = t & 63;
            float b0 = Kb[(size_t)(k0 + jp) * Wd + (n0 + n)];
            float b1 = Kb[(size_t)(k0 + jp + 1) * Wd + (n0 + n)];
            int lne = (n & 15) + 16 * ((jp >> 3) & 1);
            int e   = (jp >> 5) * 16 + (jp & 7) + 8 * ((jp >> 4) & 1);
            h2 p; p.x = (_Float16)b0; p.y = (_Float16)b1;
            *(h2*)&Bs2[(((n >> 4) * 32 + lne) * 32) + e] = p;
        }
        __syncthreads();
#pragma unroll
        for (int ck = 0; ck < 2; ++ck) {
            v16h af  = *(const v16h*)&As2[((mt * 32 + lane) * 32) + ck * 16];
            v16h bf0 = *(const v16h*)&Bs2[((ntB * 32 + lane) * 32) + ck * 16];
            v16h bf1 = *(const v16h*)&Bs2[(((ntB + 1) * 32 + lane) * 32) + ck * 16];
            acc0.v = wmma16x16x32(af, bf0, acc0.v);
            acc1.v = wmma16x16x32(af, bf1, acc1.v);
        }
        __syncthreads();
    }

    int ncol  = lane & 15;
    int mBase = m0 + mt * 16 + ((lane >> 4) ? 8 : 0);
    int nOut0 = n0 + ntB * 16 + ncol;
    int nOut1 = n0 + (ntB + 1) * 16 + ncol;
#pragma unroll
    for (int r = 0; r < 8; ++r) {
        int m = mBase + r;
        Ob[(size_t)m * Wd + nOut0] = acc0.e[r] * 0.03125f + Pb[(size_t)m * Wd + nOut0];
        Ob[(size_t)m * Wd + nOut1] = acc1.e[r] * 0.03125f + Pb[(size_t)m * Wd + nOut1];
    }
}

// ---------------------------------------------------------------------------
// 7) softmax along last axis of qk (rows of 512); 128 threads per row
// ---------------------------------------------------------------------------
__global__ void softmax_kernel(const float* __restrict__ S, float* __restrict__ A) {
    int row = blockIdx.x;
    const float* sr = S + (size_t)row * Wd;
    float* ar = A + (size_t)row * Wd;
    __shared__ float red[128];

    float m = -1e30f;
    for (int i = threadIdx.x; i < Wd; i += 128) m = fmaxf(m, sr[i]);
    red[threadIdx.x] = m; __syncthreads();
    for (int o = 64; o > 0; o >>= 1) {
        if (threadIdx.x < o) red[threadIdx.x] = fmaxf(red[threadIdx.x], red[threadIdx.x + o]);
        __syncthreads();
    }
    m = red[0]; __syncthreads();

    float s = 0.f;
    for (int i = threadIdx.x; i < Wd; i += 128) s += __expf(sr[i] - m);
    red[threadIdx.x] = s; __syncthreads();
    for (int o = 64; o > 0; o >>= 1) {
        if (threadIdx.x < o) red[threadIdx.x] += red[threadIdx.x + o];
        __syncthreads();
    }
    float inv = 1.f / red[0];

    for (int i = threadIdx.x; i < Wd; i += 128) ar[i] = __expf(sr[i] - m) * inv;
}

// ---------------------------------------------------------------------------
// 8) a*v GEMM: M=wq(512), N=d(128), K=wk(512); output back to [b,c,F,w]
//    grid = (8, 2, b*c*h)
// ---------------------------------------------------------------------------
__global__ __launch_bounds__(256) void av_gemm_kernel(
        const float* __restrict__ Aprob, const float* __restrict__ V,
        float* __restrict__ O) {
    int z  = blockIdx.z;
    int h  = z % Hc;
    int bc = z / Hc;
    int m0 = blockIdx.x * 64;
    int n0 = blockIdx.y * 64;
    const float* Ab = Aprob + (size_t)z * Wd * Wd;
    const float* Vb = V + (size_t)bc * Fc * Wd + (size_t)h * HD * Wd;
    float*       Ob = O + (size_t)bc * Fc * Wd + (size_t)h * HD * Wd;

    __shared__ __align__(32) _Float16 As2[4096];
    __shared__ __align__(32) _Float16 Bs2[4096];

    int tid = threadIdx.x, lane = tid & 31, wid = tid >> 5;
    int mt = wid & 3, ntB = (wid >> 2) * 2;

    CFrag acc0, acc1;
#pragma unroll
    for (int r = 0; r < 8; ++r) { acc0.e[r] = 0.f; acc1.e[r] = 0.f; }

    for (int k0 = 0; k0 < Wd; k0 += 64) {
        // stage A: prob matrix row-major (k contiguous)
        for (int t = tid; t < 2048; t += 256) {
            int i = t >> 5, jp = (t & 31) * 2;
            float a0 = Ab[(size_t)(m0 + i) * Wd + (k0 + jp)];
            float a1 = Ab[(size_t)(m0 + i) * Wd + (k0 + jp + 1)];
            int lne = (i & 15) + 16 * ((jp >> 3) & 1);
            int e   = (jp >> 5) * 16 + (jp & 7) + 8 * ((jp >> 4) & 1);
            h2 p; p.x = (_Float16)a0; p.y = (_Float16)a1;
            *(h2*)&As2[(((i >> 4) * 32 + lne) * 32) + e] = p;
        }
        // stage B: elem (k=kp, n) = Vb[n*Wd + k]; k contiguous pairs
        for (int t = tid; t < 2048; t += 256) {
            int n = t >> 5, kp = (t & 31) * 2;
            float b0 = Vb[(size_t)(n0 + n) * Wd + (k0 + kp)];
            float b1 = Vb[(size_t)(n0 + n) * Wd + (k0 + kp + 1)];
            int lne = (n & 15) + 16 * ((kp >> 3) & 1);
            int e   = (kp >> 5) * 16 + (kp & 7) + 8 * ((kp >> 4) & 1);
            h2 p; p.x = (_Float16)b0; p.y = (_Float16)b1;
            *(h2*)&Bs2[(((n >> 4) * 32 + lne) * 32) + e] = p;
        }
        __syncthreads();
#pragma unroll
        for (int ck = 0; ck < 2; ++ck) {
            v16h af  = *(const v16h*)&As2[((mt * 32 + lane) * 32) + ck * 16];
            v16h bf0 = *(const v16h*)&Bs2[((ntB * 32 + lane) * 32) + ck * 16];
            v16h bf1 = *(const v16h*)&Bs2[(((ntB + 1) * 32 + lane) * 32) + ck * 16];
            acc0.v = wmma16x16x32(af, bf0, acc0.v);
            acc1.v = wmma16x16x32(af, bf1, acc1.v);
        }
        __syncthreads();
    }

    // store transposed: Ob[f = h*128 + n, w = m]
    int ncol  = lane & 15;
    int mBase = m0 + mt * 16 + ((lane >> 4) ? 8 : 0);
    int nOut0 = n0 + ntB * 16 + ncol;
    int nOut1 = n0 + (ntB + 1) * 16 + ncol;
#pragma unroll
    for (int r = 0; r < 8; ++r) {
        Ob[(size_t)nOut0 * Wd + (mBase + r)] = acc0.e[r];
        Ob[(size_t)nOut1 * Wd + (mBase + r)] = acc1.e[r];
    }
}

// ---------------------------------------------------------------------------
// 9) residual add: h += z
// ---------------------------------------------------------------------------
__global__ void add_kernel(float* __restrict__ h, const float* __restrict__ z) {
    size_t idx = (size_t)blockIdx.x * blockDim.x + threadIdx.x;
    if (idx >= (size_t)Bc * Cc * Fc * Wd) return;
    h[idx] += z[idx];
}

// ---------------------------------------------------------------------------
// 10) FFN conv1: t = relu(conv3x3(hn2))^2 ; channels 8 -> 32
// ---------------------------------------------------------------------------
__global__ void ffn1_kernel(const float* __restrict__ hn2,
                            const float* __restrict__ w1,
                            const float* __restrict__ b1,
                            float* __restrict__ t) {
    size_t idx = (size_t)blockIdx.x * blockDim.x + threadIdx.x;
    if (idx >= (size_t)Bc * CEXP * Fc * Wd) return;
    int w = idx % Wd;
    int f = (idx / Wd) % Fc;
    int e = (idx / ((size_t)Wd * Fc)) % CEXP;
    int b = idx / ((size_t)Wd * Fc * CEXP);
    float acc = b1[e];
    for (int ci = 0; ci < Cc; ++ci) {
        const float* ib = hn2 + ((size_t)(b * Cc + ci)) * Fc * Wd;
        const float* wk = w1 + ((size_t)(e * Cc + ci)) * 9;
#pragma unroll
        for (int i = 0; i < 3; ++i) {
            int ff = f - 1 + i;
            if (ff < 0 || ff >= Fc) continue;
#pragma unroll
            for (int j = 0; j < 3; ++j) {
                int ww = w - 1 + j;
                if (ww < 0 || ww >= Wd) continue;
                acc += ib[(size_t)ff * Wd + ww] * wk[i * 3 + j];
            }
        }
    }
    float r = fmaxf(acc, 0.f);
    t[idx] = r * r;
}

// ---------------------------------------------------------------------------
// 11) FFN conv2 + residual, then final writes to out_h and out_concat[2..9]
// ---------------------------------------------------------------------------
__global__ void ffn2_kernel(const float* __restrict__ t,
                            const float* __restrict__ w2,
                            const float* __restrict__ b2,
                            const float* __restrict__ h,
                            float* __restrict__ out_h,
                            float* __restrict__ out_concat) {
    size_t idx = (size_t)blockIdx.x * blockDim.x + threadIdx.x;
    if (idx >= (size_t)Bc * Cc * Fc * Wd) return;
    int w = idx % Wd;
    int f = (idx / Wd) % Fc;
    int c = (idx / ((size_t)Wd * Fc)) % Cc;
    int b = idx / ((size_t)Wd * Fc * Cc);
    float acc = b2[c];
    for (int e = 0; e < CEXP; ++e) {
        const float* ib = t + ((size_t)(b * CEXP + e)) * Fc * Wd;
        const float* wk = w2 + ((size_t)(c * CEXP + e)) * 9;
#pragma unroll
        for (int i = 0; i < 3; ++i) {
            int ff = f - 1 + i;
            if (ff < 0 || ff >= Fc) continue;
#pragma unroll
            for (int j = 0; j < 3; ++j) {
                int ww = w - 1 + j;
                if (ww < 0 || ww >= Wd) continue;
                acc += ib[(size_t)ff * Wd + ww] * wk[i * 3 + j];
            }
        }
    }
    float hf = h[idx] + acc;
    out_h[idx] = hf;
    out_concat[((size_t)(b * (CINc + Cc) + 2 + c) * Fc + f) * Wd + w] = hf;
}

// ---------------------------------------------------------------------------
// launch
// ---------------------------------------------------------------------------
extern "C" void kernel_launch(void* const* d_in, const int* in_sizes, int n_in,
                              void* d_out, int out_size, void* d_ws, size_t ws_size,
                              hipStream_t stream) {
    const float* x        = (const float*)d_in[0];
    const float* prev_qk  = (const float*)d_in[1];
    const float* embed_w  = (const float*)d_in[2];
    const float* embed_b  = (const float*)d_in[3];
    const float* norm1_w  = (const float*)d_in[4];
    const float* norm1_b  = (const float*)d_in[5];
    const float* q_conv_w = (const float*)d_in[6];
    const float* q_lin_w  = (const float*)d_in[7];
    const float* k_conv_w = (const float*)d_in[8];
    const float* k_lin_w  = (const float*)d_in[9];
    const float* v_conv_w = (const float*)d_in[10];
    const float* v_lin_w  = (const float*)d_in[11];
    const float* o_lin_w  = (const float*)d_in[12];
    const float* norm2_w  = (const float*)d_in[13];
    const float* norm2_b  = (const float*)d_in[14];
    const float* conv1_w  = (const float*)d_in[15];
    const float* conv1_b  = (const float*)d_in[16];
    const float* conv2_w  = (const float*)d_in[17];
    const float* conv2_b  = (const float*)d_in[18];

    const size_t S = (size_t)Bc * Cc * Fc * Wd;      // 8,388,608
    float* out        = (float*)d_out;
    float* out_concat = out;                                      // [2,10,1024,512]
    float* out_h      = out + (size_t)Bc * (CINc + Cc) * Fc * Wd; // [2,8,1024,512]
    float* out_qk     = out_h + S;                                // [2,8,8,512,512]

    float* ws   = (float*)d_ws;
    float* hbuf = ws;            // h (embed, residual target)
    float* hn   = ws + S;        // layernorm output
    float* dwt  = ws + 2 * S;    // depthwise-conv scratch; later hn2
    float* qb   = ws + 3 * S;    // q; later reused for z
    float* kb   = ws + 4 * S;    // k
    float* vb   = ws + 5 * S;    // v
    float* obuf = ws + 6 * S;    // attention output
    float* abuf = ws + 7 * S;    // 4S region: softmax probs, later FFN hidden
    float* zb   = qb;

    dim3 blk256(256);

    // 1) embed + x copy
    embed_kernel<<<dim3((Bc * Fc * Wd + 255) / 256), blk256, 0, stream>>>(
        x, embed_w, embed_b, hbuf, out_concat);

    // 2) layernorm1
    layernorm_kernel<<<dim3(Bc * Cc * Wd), blk256, 0, stream>>>(
        hbuf, norm1_w, norm1_b, hn);

    dim3 gemm_grid(Fc / 64, Wd / 64, Bc * Cc);
    int dw_blocks = (int)((S + 255) / 256);

    // 3-5) q,k,v: depthwise 9x9 -> per-channel FxF WMMA GEMM
    dwconv9_kernel<<<dim3(dw_blocks), blk256, 0, stream>>>(hn, q_conv_w, dwt);
    mclinear_gemm_kernel<<<gemm_grid, blk256, 0, stream>>>(q_lin_w, dwt, qb);
    dwconv9_kernel<<<dim3(dw_blocks), blk256, 0, stream>>>(hn, k_conv_w, dwt);
    mclinear_gemm_kernel<<<gemm_grid, blk256, 0, stream>>>(k_lin_w, dwt, kb);
    dwconv9_kernel<<<dim3(dw_blocks), blk256, 0, stream>>>(hn, v_conv_w, dwt);
    mclinear_gemm_kernel<<<gemm_grid, blk256, 0, stream>>>(v_lin_w, dwt, vb);

    // 6) rotary on q and k
    int rot_blocks = (int)((S / 2 + 255) / 256);
    rotary_kernel<<<dim3(rot_blocks), blk256, 0, stream>>>(qb);
    rotary_kernel<<<dim3(rot_blocks), blk256, 0, stream>>>(kb);

    // 7) qk = q.k^T / 32 + prev_qk -> written directly into qk output
    qk_gemm_kernel<<<dim3(Wd / 64, Wd / 64, Bc * Cc * Hc), blk256, 0, stream>>>(
        qb, kb, prev_qk, out_qk);

    // 8) softmax
    softmax_kernel<<<dim3(Bc * Cc * Hc * Wd), dim3(128), 0, stream>>>(out_qk, abuf);

    // 9) o = a.v
    av_gemm_kernel<<<dim3(Wd / 64, HD / 64, Bc * Cc * Hc), blk256, 0, stream>>>(
        abuf, vb, obuf);

    // 10) z = o_lin(o)
    mclinear_gemm_kernel<<<gemm_grid, blk256, 0, stream>>>(o_lin_w, obuf, zb);

    // 11) h += z
    add_kernel<<<dim3(dw_blocks), blk256, 0, stream>>>(hbuf, zb);

    // 12) layernorm2 (into dwt, now hn2)
    layernorm_kernel<<<dim3(Bc * Cc * Wd), blk256, 0, stream>>>(
        hbuf, norm2_w, norm2_b, dwt);

    // 13) FFN conv1 + relu^2 (into abuf, now the 4S hidden tensor)
    ffn1_kernel<<<dim3((int)(((size_t)Bc * CEXP * Fc * Wd + 255) / 256)), blk256,
                  0, stream>>>(dwt, conv1_w, conv1_b, abuf);

    // 14) FFN conv2 + residual -> out_h and out_concat[2..9]
    ffn2_kernel<<<dim3(dw_blocks), blk256, 0, stream>>>(
        abuf, conv2_w, conv2_b, hbuf, out_h, out_concat);
}
